// TwoStageSparseCoordination_84593675862306
// MI455X (gfx1250) — matrix operand order repaired and checked by reading, not compiled, not run on hardware
//
#include <hip/hip_runtime.h>
#include <hip/hip_bf16.h>

// ---------------------------------------------------------------------------
// TwoStageSparseCoordination on MI455X (gfx1250, wave32, WMMA bf16)
// B=4 N=256 D=128 H=256 K=8
// hidden[b,i,j,h] = Apre[b,i,h] + [ s_i*s_j , s_j ](K=256) @ Wpack(K=256,H)
// ---------------------------------------------------------------------------

#define B_ 4
#define N_ 256
#define D_ 128
#define H_ 256
#define TOPK_ 8
#define LDS_ROW_U32 132        // (256 bf16 + 8 pad bf16)/2 -> 528B row stride

typedef __attribute__((ext_vector_type(16))) __bf16 v16bf;
typedef __attribute__((ext_vector_type(8)))  float  v8f;

union V16 { v16bf v; uint4 q[2]; };

__device__ __forceinline__ unsigned short f2bf(float f) {
    union { float f; unsigned u; } v; v.f = f;
    unsigned r = v.u + 0x7FFFu + ((v.u >> 16) & 1u);   // round-to-nearest-even
    return (unsigned short)(r >> 16);
}
__device__ __forceinline__ unsigned pack_bf(float a, float b) {
    return (unsigned)f2bf(a) | ((unsigned)f2bf(b) << 16);
}

// ---------------------------------------------------------------------------
// Kernel 1: Apre[b,i,h] = s[b,i,:] @ W1[0:D,:] + b1[h]   (both MLPs)
// ---------------------------------------------------------------------------
__global__ __launch_bounds__(256) void precompute_apre(
    const float* __restrict__ s,
    const float* __restrict__ Wg1, const float* __restrict__ bg1,
    const float* __restrict__ Wa1, const float* __restrict__ ba1,
    float* __restrict__ apre_g, float* __restrict__ apre_a)
{
    int blk   = blockIdx.x;
    int which = blk >> 10;          // 0 = gate, 1 = att   (B*N = 1024)
    int bi    = blk & 1023;
    int h     = threadIdx.x;
    const float* W    = which ? Wa1 : Wg1;
    const float* bias = which ? ba1 : bg1;
    const float* srow = s + (size_t)bi * D_;
    float acc = bias[h];
    #pragma unroll 8
    for (int d = 0; d < D_; ++d)
        acc = fmaf(srow[d], W[d * H_ + h], acc);
    float* out = which ? apre_a : apre_g;
    out[(size_t)bi * H_ + h] = acc;
}

// ---------------------------------------------------------------------------
// Kernel 2: bf16-pack the (K=256, H=256) weight panel per MLP into WMMA-B
// per-lane layout.  Rows 0..127 <- W1[2D+k,:], rows 128..255 <- W1[D+k,:]
// ushort index = ((mlp*8+k)*256 + h)*32 + lh*16 + t ;  source K = k*32+lh*16+t
// ---------------------------------------------------------------------------
__global__ __launch_bounds__(256) void pack_w(
    const float* __restrict__ Wg1, const float* __restrict__ Wa1,
    unsigned short* __restrict__ wpack)
{
    int idx = blockIdx.x * 256 + threadIdx.x;     // 131072 total
    int mlp = idx >> 16;
    int r   = idx & 65535;
    int t  = r & 15;
    int lh = (r >> 4) & 1;
    int h  = (r >> 5) & 255;
    int k  = r >> 13;
    int kk = k * 32 + lh * 16 + t;                // 0..255 within K-ext
    int row = (kk < D_) ? (2 * D_ + kk) : (D_ + (kk - D_));
    const float* W = mlp ? Wa1 : Wg1;
    wpack[idx] = f2bf(W[row * H_ + h]);
}

// ---------------------------------------------------------------------------
// Kernel 3: fused pair-MLP x2 + top-k gate + masked softmax + context
// one workgroup (256 thr = 8 waves) per (b,i)
// ---------------------------------------------------------------------------
__global__ __launch_bounds__(256) void fused_pair(
    const float* __restrict__ s,
    const float* __restrict__ wg2, const float* __restrict__ wa2,
    const float* __restrict__ apre_g, const float* __restrict__ apre_a,
    const unsigned short* __restrict__ wpack,
    float* __restrict__ out)
{
    __shared__ unsigned int spr[N_ * LDS_ROW_U32];   // 135168 B K-panel (bf16)
    __shared__ float sc_g[N_];
    __shared__ float sc_a[N_];
    __shared__ float si[D_];
    __shared__ float ap_sh[2][H_];                   // Apre rows (g,a)
    __shared__ float w2_sh[2][H_];                   // W2 columns (g,a)
    __shared__ int   sel_idx_sh[TOPK_];
    __shared__ float sel_w_sh[TOPK_];

    const int tid = threadIdx.x;
    const int bi  = blockIdx.x;                      // b*N + i
    const float* srow_i = s + (size_t)bi * D_;
    if (tid < D_) si[tid] = srow_i[tid];
    ap_sh[0][tid] = apre_g[(size_t)bi * H_ + tid];
    ap_sh[1][tid] = apre_a[(size_t)bi * H_ + tid];
    w2_sh[0][tid] = wg2[tid];
    w2_sh[1][tid] = wa2[tid];
    __syncthreads();

    // ---- phase 1: build K-panel row j=tid : [ s_j*s_i (128) | s_j (128) ] bf16
    {
        const float* srow_j = s + ((size_t)(bi & ~255) + tid) * D_;  // s[b, tid, :]
        unsigned int* dst = spr + tid * LDS_ROW_U32;
        #pragma unroll 8
        for (int d2 = 0; d2 < D_ / 2; ++d2) {
            float2 v  = ((const float2*)srow_j)[d2];
            float2 sv = ((const float2*)si)[d2];
            dst[d2]      = pack_bf(v.x * sv.x, v.y * sv.y);
            dst[64 + d2] = pack_bf(v.x, v.y);
        }
    }
    __syncthreads();

    // ---- phase 2: hidden = panel @ Wpack, fused ReLU + dot(W2) epilogue
    const int lane = tid & 31;
    const int w    = tid >> 5;        // wave id 0..7
    const int lh   = lane >> 4;       // half-wave
    const int ln   = lane & 15;
    const int jt0  = w * 2;           // this wave owns j-tiles jt0, jt0+1

    const uint4* wp4 = (const uint4*)wpack;

    #pragma unroll 1
    for (int jtl = 0; jtl < 2; ++jtl) {
        const int jt = jt0 + jtl;
        // A-operand cache for this j-tile: 8 k-chunks = 64 VGPRs
        const unsigned int* rbase = spr + (jt * 16 + ln) * LDS_ROW_U32;
        V16 aop[8];
        #pragma unroll
        for (int k = 0; k < 8; ++k) {
            const unsigned int* p = rbase + 16 * k + 8 * lh;
            aop[k].q[0] = *(const uint4*)(p);
            aop[k].q[1] = *(const uint4*)(p + 4);
        }
        float spart[2][8];            // [mlp][r] partial scores, this j-tile
        #pragma unroll
        for (int a1 = 0; a1 < 2; ++a1)
            #pragma unroll
            for (int a2 = 0; a2 < 8; ++a2) spart[a1][a2] = 0.f;

        #pragma unroll 1
        for (int ht = 0; ht < 16; ++ht) {
            const int h = ht * 16 + ln;
            #pragma unroll
            for (int mlp = 0; mlp < 2; ++mlp) {      // FULL unroll: spart stays in VGPRs
                const float ap  = ap_sh[mlp][h];
                const float w2v = w2_sh[mlp][h];
                v8f acc = {};
                #pragma unroll
                for (int k = 0; k < 8; ++k) {
                    V16 bop;
                    size_t q = ((((size_t)mlp * 8 + k) * 256 + h) * 2 + lh) * 2;
                    bop.q[0] = wp4[q];
                    bop.q[1] = wp4[q + 1];
                    acc = __builtin_amdgcn_wmma_f32_16x16x32_bf16(
                        false, aop[k].v, false, bop.v, (short)0, acc, false, false);
                }
                #pragma unroll
                for (int r = 0; r < 8; ++r) {
                    float hid = fmaxf(acc[r] + ap, 0.f);
                    spart[mlp][r] = fmaf(hid, w2v, spart[mlp][r]);
                }
            }
        }

        // reduce partial scores across the 16 h-lanes of each half-wave
        #pragma unroll
        for (int mlp = 0; mlp < 2; ++mlp)
            #pragma unroll
            for (int r = 0; r < 8; ++r) {
                float v = spart[mlp][r];
                v += __shfl_xor(v, 1, 16);
                v += __shfl_xor(v, 2, 16);
                v += __shfl_xor(v, 4, 16);
                v += __shfl_xor(v, 8, 16);
                if (ln == 0) {
                    int j = jt * 16 + r + lh * 8;    // C-tile row mapping
                    (mlp ? sc_a : sc_g)[j] = v;
                }
            }
    }
    __syncthreads();

    // ---- phase 3: top-K over sc_g + softmax of sc_a on the selected set
    if (tid < 32) {
        float vals[8];
        #pragma unroll
        for (int c = 0; c < 8; ++c) vals[c] = sc_g[lane * 8 + c];
        int myj = 0;
        for (int it = 0; it < TOPK_; ++it) {
            float bv = -__builtin_inff(); int bc = 0;
            #pragma unroll
            for (int c = 0; c < 8; ++c)
                if (vals[c] > bv) { bv = vals[c]; bc = c; }
            int bidx = lane * 8 + bc;
            #pragma unroll
            for (int o = 16; o >= 1; o >>= 1) {
                float ov = __shfl_xor(bv, o, 32);
                int   oi = __shfl_xor(bidx, o, 32);
                if (ov > bv || (ov == bv && oi < bidx)) { bv = ov; bidx = oi; }
            }
            if (lane == it) myj = bidx;
            // constant-indexed invalidation (avoid dynamic reg-array index)
            #pragma unroll
            for (int c = 0; c < 8; ++c)
                if ((bidx >> 3) == lane && (bidx & 7) == c)
                    vals[c] = -__builtin_inff();
            if (lane == 0) sel_idx_sh[it] = bidx;
        }
        float la = (lane < TOPK_) ? sc_a[myj] : -__builtin_inff();
        float m = la;
        #pragma unroll
        for (int o = 16; o >= 1; o >>= 1) m = fmaxf(m, __shfl_xor(m, o, 32));
        float e = (lane < TOPK_) ? __expf(la - m) : 0.f;
        float z = e;
        #pragma unroll
        for (int o = 16; o >= 1; o >>= 1) z += __shfl_xor(z, o, 32);
        if (lane < TOPK_) sel_w_sh[lane] = e / z;
    }
    __syncthreads();

    // ---- phase 4: outputs (ctx | gate | att_weights, concatenated)
    const size_t gate_base = (size_t)B_ * N_ * D_;                 // 131072
    const size_t attw_base = gate_base + (size_t)B_ * N_ * N_;     // 393216
    {
        int j = tid;
        bool issel = false; float wv = 0.f;
        #pragma unroll
        for (int c = 0; c < TOPK_; ++c)
            if (sel_idx_sh[c] == j) { issel = true; wv = sel_w_sh[c]; }
        size_t rowoff = (size_t)bi * N_ + j;
        out[gate_base + rowoff] = issel ? 1.f : 0.f;
        out[attw_base + rowoff] = issel ? wv : 0.f;
    }
    if (tid < D_) {
        float acc = 0.f;
        #pragma unroll
        for (int c = 0; c < TOPK_; ++c)
            acc += sel_w_sh[c] * s[((size_t)(bi & ~255) + sel_idx_sh[c]) * D_ + tid];
        out[(size_t)bi * D_ + tid] = acc;
    }
}

// ---------------------------------------------------------------------------
extern "C" void kernel_launch(void* const* d_in, const int* in_sizes, int n_in,
                              void* d_out, int out_size, void* d_ws, size_t ws_size,
                              hipStream_t stream) {
    (void)in_sizes; (void)n_in; (void)out_size; (void)ws_size;
    const float* s   = (const float*)d_in[0];
    const float* Wg1 = (const float*)d_in[1];
    const float* bg1 = (const float*)d_in[2];
    const float* Wg2 = (const float*)d_in[3];
    const float* Wa1 = (const float*)d_in[5];
    const float* ba1 = (const float*)d_in[6];
    const float* Wa2 = (const float*)d_in[7];
    float* out = (float*)d_out;

    const size_t APRE_ELEMS = (size_t)B_ * N_ * H_;   // 262144 floats each
    float* apre_g = (float*)d_ws;
    float* apre_a = apre_g + APRE_ELEMS;
    unsigned short* wpack = (unsigned short*)(apre_a + APRE_ELEMS); // 131072 bf16

    precompute_apre<<<2 * B_ * N_, 256, 0, stream>>>(s, Wg1, bg1, Wa1, ba1,
                                                     apre_g, apre_a);
    pack_w<<<512, 256, 0, stream>>>(Wg1, Wa1, wpack);
    fused_pair<<<B_ * N_, 256, 0, stream>>>(s, Wg2, Wa2, apre_g, apre_a,
                                            wpack, out);
}
